// GraphAttentionLayer_70815420776683
// MI455X (gfx1250) — matrix operand order repaired and checked by reading, not compiled, not run on hardware
//
#include <hip/hip_runtime.h>
#include <hip/hip_bf16.h>

typedef __attribute__((ext_vector_type(16))) _Float16 v16h;
typedef __attribute__((ext_vector_type(8)))  _Float16 v8h;
typedef __attribute__((ext_vector_type(4)))  _Float16 v4h;
typedef __attribute__((ext_vector_type(8)))  float    v8f;

#define N_NODES 8192
#define F_IN    256
#define F_OUT   128
#define LRELU_A 0.2f
#define P_PAD   40   // LDS row stride (halves) for the P tile, padded vs 32 to spread banks
#define NTILES  (N_NODES / 32)

// ---------------------------------------------------------------------------
// Kernel 0: init global max slot to -inf
// ---------------------------------------------------------------------------
__global__ void gat_init(float* gmax) {
    if (threadIdx.x == 0 && blockIdx.x == 0) *gmax = -__builtin_inff();
}

// ---------------------------------------------------------------------------
// Kernel 1: Wh = h @ W   (8192x256 @ 256x128, fp32 in -> f16 WMMA -> fp32 out)
// grid = 512 blocks (16 rows each), block = 256 threads (8 waves; wave w owns
// the 16-column F_out tile [16w, 16w+16)).
// A k order used here (contiguous, same as B): lane l (m=l&15, kh=l>>4): half j = K(kh*16+j).
// B layout: lane l (n=l&15, kg=l>>4): half j = K(kg*16+j).
// ---------------------------------------------------------------------------
__global__ __launch_bounds__(256) void gat_wh_gemm(const float* __restrict__ h,
                                                   const float* __restrict__ W,
                                                   float* __restrict__ Wh) {
    const int i0   = blockIdx.x * 16;
    const int wave = threadIdx.x >> 5;
    const int lane = threadIdx.x & 31;
    const int m    = lane & 15;
    const int kh   = lane >> 4;
    const int f0   = wave * 16;

    v8f acc = {};
    for (int kk = 0; kk < F_IN; kk += 32) {
        // A fragment from h (row-major)
        const float* hrow = h + (size_t)(i0 + m) * F_IN + kk + kh * 16;
        v16h a;
#pragma unroll
        for (int j = 0; j < 8; ++j) {
            a[j]     = (_Float16)hrow[j];
            a[j + 8] = (_Float16)hrow[8 + j];
        }
        // B fragment from W (row-major, column-strided reads)
        const float* wcol = W + (size_t)(kk + kh * 16) * F_OUT + f0 + m;
        v16h b;
#pragma unroll
        for (int j = 0; j < 16; ++j) b[j] = (_Float16)wcol[(size_t)j * F_OUT];

        acc = __builtin_amdgcn_wmma_f32_16x16x32_f16(false, a, false, b,
                                                     (short)0, acc, false, false);
    }
#pragma unroll
    for (int r = 0; r < 8; ++r)
        Wh[(size_t)(i0 + r + 8 * kh) * F_OUT + f0 + m] = acc[r];
}

// ---------------------------------------------------------------------------
// Kernel 2: Wh1 = Wh@a[:128], Wh2 = Wh@a[128:], gmax = max(Wh2) (float atomic)
// one thread per row; grid 32 x 256
// ---------------------------------------------------------------------------
__global__ __launch_bounds__(256) void gat_vecs(const float* __restrict__ Wh,
                                                const float* __restrict__ a,
                                                float* __restrict__ Wh1,
                                                float* __restrict__ Wh2,
                                                float* gmax) {
    const int i = blockIdx.x * blockDim.x + threadIdx.x;
    const float* row = Wh + (size_t)i * F_OUT;
    float s1 = 0.f, s2 = 0.f;
#pragma unroll 4
    for (int f = 0; f < F_OUT; ++f) {
        float v = row[f];
        s1 += v * a[f];
        s2 += v * a[F_OUT + f];
    }
    Wh1[i] = s1;
    Wh2[i] = s2;

    // wave32 max-reduce then one atomic per wave (float-max via int trick)
    float mx = s2;
#pragma unroll
    for (int off = 16; off > 0; off >>= 1)
        mx = fmaxf(mx, __shfl_xor(mx, off, 32));
    if ((threadIdx.x & 31) == 0) {
        if (mx >= 0.f) atomicMax((int*)gmax, __float_as_int(mx));
        else           atomicMin((unsigned int*)gmax, __float_as_uint(mx));
    }
}

// ---------------------------------------------------------------------------
// Kernel 3: repack Wh (fp32) into f16 WMMA-B fragment layout so the main
// kernel's B load is one contiguous 32B/lane (fully coalesced) read.
// WhB[((jb*8 + fb)*32 + lane)*16 + j] = Wh[jb*32 + (lane>>4)*16 + j][fb*16 + (lane&15)]
// ---------------------------------------------------------------------------
__global__ __launch_bounds__(256) void gat_repack(const float* __restrict__ Wh,
                                                  _Float16* __restrict__ WhB) {
    const int t    = blockIdx.x * 256 + threadIdx.x;   // [0, 65536)
    const int lane = t & 31;
    const int fb   = (t >> 5) & 7;
    const int jb   = t >> 8;
    const int n    = lane & 15;
    const int kg   = lane >> 4;
    const float* src = Wh + (size_t)(jb * 32 + kg * 16) * F_OUT + fb * 16 + n;
    v16h b;
#pragma unroll
    for (int j = 0; j < 16; ++j) b[j] = (_Float16)src[(size_t)j * F_OUT];
    *(v16h*)(WhB + (size_t)t * 16) = b;
}

// ---------------------------------------------------------------------------
// Kernel 4: fused masked-softmax + attention@Wh + ELU (flash style, one pass
// over adj). grid = 256 blocks (32 rows each), block = 256 threads (8 waves).
//
// Software-pipelined with a double-buffered P tile:
//   iter jt: issue adj/Wh2 loads for tile jt+1  -> WMMA consumes sP[jt&1]
//            -> exp/mask/reduce writes sP[(jt+1)&1] -> ONE barrier.
// M_i = lrelu(Wh1_i + gmax) is a softmax-safe upper bound on the row max
// (lrelu is monotone), so no online-max rescaling is needed.
// ---------------------------------------------------------------------------
__global__ __launch_bounds__(256) void gat_main(const int* __restrict__ adj,
                                                const float* __restrict__ Wh1,
                                                const float* __restrict__ Wh2,
                                                const float* __restrict__ gmax,
                                                const _Float16* __restrict__ WhB,
                                                float* __restrict__ out) {
    __shared__ float    sW1[32];
    __shared__ float    sM[32];
    __shared__ float    sL[32];
    __shared__ _Float16 sP[2][32 * P_PAD];

    const int i0 = blockIdx.x * 32;

    if (threadIdx.x < 32) {
        float w1 = Wh1[i0 + threadIdx.x];
        float e  = w1 + *gmax;
        sW1[threadIdx.x] = w1;
        sM[threadIdx.x]  = (e > 0.f) ? e : LRELU_A * e;   // upper bound M_i
    }
    __syncthreads();

    // phase-1 indexing: 8 threads per row, 4 columns per thread
    const int row = threadIdx.x >> 3;
    const int c   = threadIdx.x & 7;
    const int* adjrow = adj + (size_t)(i0 + row) * N_NODES + 4 * c;
    const float* wh2p = Wh2 + 4 * c;
    const float w1 = sW1[row];
    const float Mi = sM[row];

    // phase-2 indexing: wave w owns F_out tile [16w,16w+16), two row tiles
    const int wave = threadIdx.x >> 5;
    const int lane = threadIdx.x & 31;
    const int m    = lane & 15;
    const int hb   = lane >> 4;

    v8f acc0 = {}, acc1 = {};
    float lacc = 0.f;

    // ---- prologue: build P tile 0 into sP[0] ----
    {
        const int4   ad = *(const int4*)(adjrow);
        const float4 w2 = *(const float4*)(wh2p);
        float e0 = w1 + w2.x; e0 = (e0 > 0.f) ? e0 : LRELU_A * e0;
        float e1 = w1 + w2.y; e1 = (e1 > 0.f) ? e1 : LRELU_A * e1;
        float e2 = w1 + w2.z; e2 = (e2 > 0.f) ? e2 : LRELU_A * e2;
        float e3 = w1 + w2.w; e3 = (e3 > 0.f) ? e3 : LRELU_A * e3;
        float p0 = (ad.x > 0) ? __expf(e0 - Mi) : 0.f;
        float p1 = (ad.y > 0) ? __expf(e1 - Mi) : 0.f;
        float p2 = (ad.z > 0) ? __expf(e2 - Mi) : 0.f;
        float p3 = (ad.w > 0) ? __expf(e3 - Mi) : 0.f;
        float ps = p0 + p1 + p2 + p3;
        ps += __shfl_xor(ps, 1, 32);
        ps += __shfl_xor(ps, 2, 32);
        ps += __shfl_xor(ps, 4, 32);
        if (c == 0) lacc += ps;
        v4h pv;
        pv[0] = (_Float16)p0; pv[1] = (_Float16)p1;
        pv[2] = (_Float16)p2; pv[3] = (_Float16)p3;
        *(v4h*)(&sP[0][row * P_PAD + 4 * c]) = pv;
    }
    __syncthreads();

    for (int jt = 0; jt < NTILES; ++jt) {
        const int buf = jt & 1;

        // ---- issue next tile's global loads first (overlap with WMMA) ----
        int4   ad = {0, 0, 0, 0};
        float4 w2 = {0.f, 0.f, 0.f, 0.f};
        const bool have_next = (jt + 1) < NTILES;
        if (have_next) {
            ad = *(const int4*)(adjrow + (size_t)(jt + 1) * 32);
            w2 = *(const float4*)(wh2p + (size_t)(jt + 1) * 32);
            if (jt + 2 < NTILES)
                __builtin_prefetch(adjrow + (size_t)(jt + 2) * 32, 0, 3);
        }

        // ---- phase 2: acc += P @ Wh[j-tile, f-tile] via WMMA on sP[buf] ----
        {
            const _Float16* pr0 = &sP[buf][m * P_PAD];
            const _Float16* pr1 = &sP[buf][(16 + m) * P_PAD];
            v8h lo0 = *(const v8h*)(pr0 + hb * 16);
            v8h hi0 = *(const v8h*)(pr0 + hb * 16 + 8);
            v8h lo1 = *(const v8h*)(pr1 + hb * 16);
            v8h hi1 = *(const v8h*)(pr1 + hb * 16 + 8);
            v16h a0, a1;
#pragma unroll
            for (int j = 0; j < 8; ++j) {
                a0[j] = lo0[j]; a0[j + 8] = hi0[j];
                a1[j] = lo1[j]; a1[j + 8] = hi1[j];
            }
            v16h b = *(const v16h*)(WhB + ((size_t)(jt * 8 + wave) * 32 + lane) * 16);

            acc0 = __builtin_amdgcn_wmma_f32_16x16x32_f16(false, a0, false, b,
                                                          (short)0, acc0, false, false);
            acc1 = __builtin_amdgcn_wmma_f32_16x16x32_f16(false, a1, false, b,
                                                          (short)0, acc1, false, false);
        }

        // ---- phase 1 for tile jt+1 into the other buffer ----
        if (have_next) {
            float e0 = w1 + w2.x; e0 = (e0 > 0.f) ? e0 : LRELU_A * e0;
            float e1 = w1 + w2.y; e1 = (e1 > 0.f) ? e1 : LRELU_A * e1;
            float e2 = w1 + w2.z; e2 = (e2 > 0.f) ? e2 : LRELU_A * e2;
            float e3 = w1 + w2.w; e3 = (e3 > 0.f) ? e3 : LRELU_A * e3;
            float p0 = (ad.x > 0) ? __expf(e0 - Mi) : 0.f;
            float p1 = (ad.y > 0) ? __expf(e1 - Mi) : 0.f;
            float p2 = (ad.z > 0) ? __expf(e2 - Mi) : 0.f;
            float p3 = (ad.w > 0) ? __expf(e3 - Mi) : 0.f;
            float ps = p0 + p1 + p2 + p3;
            ps += __shfl_xor(ps, 1, 32);
            ps += __shfl_xor(ps, 2, 32);
            ps += __shfl_xor(ps, 4, 32);
            if (c == 0) lacc += ps;
            v4h pv;
            pv[0] = (_Float16)p0; pv[1] = (_Float16)p1;
            pv[2] = (_Float16)p2; pv[3] = (_Float16)p3;
            *(v4h*)(&sP[buf ^ 1][row * P_PAD + 4 * c]) = pv;
        }

        // one barrier per tile: publishes sP[buf^1] for the next iteration and
        // guarantees this iteration's reads of sP[buf] are done before iter
        // jt+1 overwrites it (writes to sP[buf] happen after the NEXT barrier).
        __syncthreads();
    }

    if (c == 0) sL[row] = lacc;
    __syncthreads();

    // epilogue: divide by row sum, ELU, store fp32
    const int f0 = wave * 16;
#pragma unroll
    for (int r = 0; r < 8; ++r) {
        int r0 = r + 8 * hb;
        float x0 = acc0[r] / (sL[r0] + 1e-30f);
        out[(size_t)(i0 + r0) * F_OUT + f0 + m] = (x0 > 0.f) ? x0 : __expf(x0) - 1.f;
        int r1 = 16 + r + 8 * hb;
        float x1 = acc1[r] / (sL[r1] + 1e-30f);
        out[(size_t)(i0 + r1) * F_OUT + f0 + m] = (x1 > 0.f) ? x1 : __expf(x1) - 1.f;
    }
}

// ---------------------------------------------------------------------------
extern "C" void kernel_launch(void* const* d_in, const int* in_sizes, int n_in,
                              void* d_out, int out_size, void* d_ws, size_t ws_size,
                              hipStream_t stream) {
    const float* h   = (const float*)d_in[0];   // [8192,256]
    const int*   adj = (const int*)  d_in[1];   // [8192,8192]
    const float* W   = (const float*)d_in[2];   // [256,128]
    const float* a   = (const float*)d_in[3];   // [256,1]
    float* out = (float*)d_out;                 // [8192,128]

    // workspace carve-up (~6.07 MB)
    char* ws = (char*)d_ws;
    float*    Wh   = (float*)ws;                               // 4 MB
    _Float16* WhB  = (_Float16*)(ws + ((size_t)4 << 20));      // 2 MB
    float*    Wh1  = (float*)(ws + ((size_t)6 << 20));         // 32 KB
    float*    Wh2  = Wh1 + N_NODES;                            // 32 KB
    float*    gmax = Wh2 + N_NODES;                            // 4 B

    gat_init  <<<1,   1,   0, stream>>>(gmax);
    gat_wh_gemm<<<N_NODES / 16, 256, 0, stream>>>(h, W, Wh);
    gat_vecs  <<<N_NODES / 256, 256, 0, stream>>>(Wh, a, Wh1, Wh2, gmax);
    gat_repack<<<(N_NODES / 32) * 8 * 32 / 256, 256, 0, stream>>>(Wh, WhB);
    gat_main  <<<N_NODES / 32, 256, 0, stream>>>(adj, Wh1, Wh2, gmax, WhB, out);
}
